// CudaMixedBitLinear_52810917871714
// MI455X (gfx1250) — compile-verified, matrix operand order checked by reading, not run on hardware
//
#include <hip/hip_runtime.h>

// W4A16 group-quantized GEMM for MI455X (gfx1250, wave32, WMMA).
// out[M, N] = x[M, K] @ dequant(qweight)[N, K]^T, fp32 accum.
// M = 4096 tokens, K = 4096, N = 11008, group = 128.
//
// - 128x128 block tile, K-stage = 128 (one quant group), 8 wave32s (2M x 4N),
//   wave tile 64x32 -> 4 Mfrags x 2 Nfrags, v_wmma_f32_16x16x32_f16.
// - X tile: global_load_async_to_lds_b128 (ASYNCcnt), double-buffered LDS.
// - W tile: global->VGPR, dequantized ONCE per block into an fp16 LDS tile
//   (fp16(0x6400|nib) == 1024+nib, one v_pk_fma_f16 per element pair with
//   bias (-zero-1024)*scale), then the inner loop is a pure LDS->WMMA GEMM.

#define IN_F   4096
#define OUT_F  11008
#define NGROUP 32          // IN_F / 128
#define BM     128
#define BN     128
#define BK     128         // one quant group per K-stage
#define LDS_STRIDE 136     // halves per row (128 + 8 pad), 272B -> conflict-free

typedef __attribute__((ext_vector_type(16))) _Float16     v16h;
typedef __attribute__((ext_vector_type(8)))  _Float16     v8h;
typedef __attribute__((ext_vector_type(2)))  _Float16     h2;
typedef __attribute__((ext_vector_type(8)))  float        v8f;
typedef __attribute__((ext_vector_type(4)))  unsigned int v4u;
typedef __attribute__((ext_vector_type(4)))  int          v4i;

#define GAS __attribute__((address_space(1)))
#define LAS __attribute__((address_space(3)))

#if __has_builtin(__builtin_amdgcn_global_load_async_to_lds_b128)
#define HAVE_ASYNC_LDS 1
#endif

static __device__ __forceinline__ void wait_async0() {
#if __has_builtin(__builtin_amdgcn_s_wait_asynccnt)
    __builtin_amdgcn_s_wait_asynccnt(0);
#else
    asm volatile("s_wait_asynccnt 0" ::: "memory");
#endif
}

__global__ __launch_bounds__(256)
void w4a16_wmma_gemm(const _Float16* __restrict__ x,
                     const int*      __restrict__ qw,
                     const _Float16* __restrict__ scales,
                     const int*      __restrict__ qzeros,
                     float*          __restrict__ out)
{
    __shared__ _Float16 Xs[2][BM * LDS_STRIDE];   // 2 x 34816 B, activations
    __shared__ _Float16 Wd[2][BN * LDS_STRIDE];   // 2 x 34816 B, dequantized W

    const unsigned tid  = threadIdx.x;
    const unsigned lane = tid & 31u;
    const unsigned wave = tid >> 5;      // 0..7
    const unsigned wm   = wave & 1u;     // 0..1  -> 64 rows of M each
    const unsigned wn   = wave >> 1;     // 0..3  -> 32 cols of N each
    const unsigned h    = lane >> 4;     // lane half (WMMA layout)
    const unsigned l16  = lane & 15u;

    const unsigned oBase = blockIdx.x * BN;
    const unsigned mBase = blockIdx.y * BM;

    v8f acc[4][2];
#pragma unroll
    for (int mf = 0; mf < 4; ++mf)
#pragma unroll
        for (int f = 0; f < 2; ++f)
            acc[mf][f] = (v8f)0.0f;

    // Stage geometry: chunk i -> idx = tid + 256*i, r = idx>>4, c = idx&15.
    // X chunk: halves [r][c*8 .. +7]; W chunk: ints [r][c*4 .. +3] -> 8 fp16.

    // ---- Persistent stage registers for the W path --------------------
    v4u      wPack[8];
    _Float16 wScale[8];
    int      wZero[8];

    auto issue_w = [&](unsigned kb) {
        const unsigned kOffI = kb * (BK / 2);
#pragma unroll
        for (unsigned i = 0; i < 8; ++i) {
            const unsigned idx = tid + 256u * i;
            const unsigned r = idx >> 4;
            const unsigned c = idx & 15u;
            const unsigned o = oBase + r;
            wPack[i]  = *(const v4u*)(qw + (size_t)o * (IN_F / 2) + kOffI + c * 4u);
            wScale[i] = scales[(size_t)o * NGROUP + kb];
            wZero[i]  = qzeros[(size_t)o * (NGROUP / 2) + (kb >> 1)];
        }
    };

    // Dequantize this thread's 32 packed bytes -> 64 fp16, store to Wd[buf].
    auto dequant_commit = [&](unsigned kb, unsigned buf) {
#pragma unroll
        for (unsigned i = 0; i < 8; ++i) {
            const unsigned idx = tid + 256u * i;
            const unsigned r = idx >> 4;
            const unsigned c = idx & 15u;
            const float s = (float)wScale[i];
            const int   z = (wZero[i] >> ((kb & 1u) * 4u)) & 15;
            h2 s2, b2;
            const _Float16 hs = (_Float16)s;
            const _Float16 hb = (_Float16)(-s * (float)(z + 1024));
            s2[0] = hs; s2[1] = hs;
            b2[0] = hb; b2[1] = hb;
            v4u dq;
#pragma unroll
            for (unsigned j = 0; j < 4; ++j) {
                const unsigned e = wPack[i][j];
                const unsigned p = 0x64006400u | (e & 0xFu) | ((e & 0xF0u) << 12);
                dq[j] = __builtin_bit_cast(unsigned,
                    __builtin_elementwise_fma(__builtin_bit_cast(h2, p), s2, b2));
            }
            *(v4u*)&Wd[buf][r * LDS_STRIDE + c * 8u] = dq;
        }
    };

#if defined(HAVE_ASYNC_LDS)
    auto issue_x = [&](unsigned kb, unsigned buf) {
        const unsigned kOffH = kb * BK;
#pragma unroll
        for (unsigned i = 0; i < 8; ++i) {
            const unsigned idx = tid + 256u * i;
            const unsigned r = idx >> 4;
            const unsigned c = idx & 15u;
            const void* gx = (const void*)(x + (size_t)(mBase + r) * IN_F + kOffH + c * 8u);
            GAS v4i* gxa = (GAS v4i*)(unsigned long long)(size_t)gx;
            LAS v4i* lxa = (LAS v4i*)(unsigned)(size_t)(const void*)&Xs[buf][r * LDS_STRIDE + c * 8u];
            __builtin_amdgcn_global_load_async_to_lds_b128(gxa, lxa, 0, 0);
        }
    };
    issue_x(0u, 0u);
#else
    v4u xReg[8];
    auto issue_x_loads = [&](unsigned kb) {
        const unsigned kOffH = kb * BK;
#pragma unroll
        for (unsigned i = 0; i < 8; ++i) {
            const unsigned idx = tid + 256u * i;
            const unsigned r = idx >> 4;
            const unsigned c = idx & 15u;
            xReg[i] = *(const v4u*)(x + (size_t)(mBase + r) * IN_F + kOffH + c * 8u);
        }
    };
    auto commit_x = [&](unsigned buf) {
#pragma unroll
        for (unsigned i = 0; i < 8; ++i) {
            const unsigned idx = tid + 256u * i;
            const unsigned r = idx >> 4;
            const unsigned c = idx & 15u;
            *(v4u*)&Xs[buf][r * LDS_STRIDE + c * 8u] = xReg[i];
        }
    };
    issue_x_loads(0u);
#endif
    issue_w(0u);

    for (unsigned kb = 0; kb < NGROUP; ++kb) {
        const unsigned buf = kb & 1u;

        // Stage-kb weights: registers -> fp16 LDS tile (overlaps other waves'
        // trailing WMMAs of the previous stage).
        dequant_commit(kb, buf);
#if !defined(HAVE_ASYNC_LDS)
        commit_x(buf);
#endif

#if defined(HAVE_ASYNC_LDS)
        wait_async0();          // this wave's stage-kb X DMAs have landed
#endif
        __syncthreads();        // Xs[buf] + Wd[buf] complete block-wide

        if (kb + 1 < NGROUP) {
#if defined(HAVE_ASYNC_LDS)
            issue_x(kb + 1, buf ^ 1u);
#else
            issue_x_loads(kb + 1);
#endif
            issue_w(kb + 1);
        }

        // ---- Pure LDS -> WMMA inner loop: 4 K-steps of 32 ----------------
#pragma unroll
        for (unsigned ks = 0; ks < 4; ++ks) {
            const unsigned kl = ks * 32u;

            v16h a[4];
#pragma unroll
            for (int mf = 0; mf < 4; ++mf) {
                const unsigned row = wm * 64u + (unsigned)mf * 16u + l16;
                const v8h* p = (const v8h*)&Xs[buf][row * LDS_STRIDE + kl + h * 8u];
                const v8h lo = p[0];
                const v8h hi = p[2];   // +16 halves
                a[mf] = __builtin_shufflevector(lo, hi,
                        0, 1, 2, 3, 4, 5, 6, 7, 8, 9, 10, 11, 12, 13, 14, 15);
            }

            v16h b[2];
#pragma unroll
            for (int f = 0; f < 2; ++f) {
                const unsigned ol = wn * 32u + (unsigned)f * 16u + l16;
                const v8h* p = (const v8h*)&Wd[buf][ol * LDS_STRIDE + kl + h * 8u];
                const v8h lo = p[0];
                const v8h hi = p[2];
                b[f] = __builtin_shufflevector(lo, hi,
                        0, 1, 2, 3, 4, 5, 6, 7, 8, 9, 10, 11, 12, 13, 14, 15);
            }

#pragma unroll
            for (int mf = 0; mf < 4; ++mf)
#pragma unroll
                for (int f = 0; f < 2; ++f)
                    acc[mf][f] = __builtin_amdgcn_wmma_f32_16x16x32_f16(
                        false, a[mf], false, b[f],
                        (short)0, acc[mf][f], false, false);
        }
    }

    // ---- Writeback: C/D layout — VGPR r -> M = r + 8*h, lane -> N = l16 ----
#pragma unroll
    for (int mf = 0; mf < 4; ++mf) {
#pragma unroll
        for (int f = 0; f < 2; ++f) {
            const unsigned o = oBase + wn * 32u + (unsigned)f * 16u + l16;
#pragma unroll
            for (int r = 0; r < 8; ++r) {
                const unsigned m = mBase + wm * 64u + (unsigned)mf * 16u + (unsigned)r + 8u * h;
                out[(size_t)m * OUT_F + o] = acc[mf][f][r];
            }
        }
    }
}

extern "C" void kernel_launch(void* const* d_in, const int* in_sizes, int n_in,
                              void* d_out, int out_size, void* d_ws, size_t ws_size,
                              hipStream_t stream)
{
    const _Float16* x      = (const _Float16*)d_in[0];
    const int*      qw     = (const int*)d_in[1];
    const _Float16* scales = (const _Float16*)d_in[2];
    const int*      qzeros = (const int*)d_in[3];
    float*          out    = (float*)d_out;

    const int M = in_sizes[0] / IN_F;           // 4096 tokens
    dim3 grid(OUT_F / BN, M / BM);              // (86, 32)
    dim3 block(256);
    w4a16_wmma_gemm<<<grid, block, 0, stream>>>(x, qw, scales, qzeros, out);
}
